// ScaledDotProductAttention_58196806861275
// MI455X (gfx1250) — matrix operand order repaired and checked by reading, not compiled, not run on hardware
//
#include <hip/hip_runtime.h>
#include <cstdint>
#include <cstddef>

// ---------------------------------------------------------------------------
// Flash-attention (2-pass, bf16 WMMA, async-LDS V staging, base-2 softmax)
// for B=4,H=16,S=2048,D=64, fp32 in/out.
// Outputs: O [B,H,S,D] then ATTN [B,H,S,S], concatenated in d_out.
// ---------------------------------------------------------------------------

typedef __bf16 bf16_t;
typedef bf16_t v16bf __attribute__((ext_vector_type(16)));
typedef float  v8f   __attribute__((ext_vector_type(8)));

#define S_LEN  2048
#define D_DIM  64
#define NBH    64            // B*H
#define QTILE  64            // q rows per workgroup
#define WAVES  4
#define BLOCK  128
#define VTS    68            // padded row stride of f32 V tile (floats; 272B, 16B-aligned)
#define PSTR   40            // padded row stride of P tile (bf16)
// 1/temperature * log2(e): scores come out in log2 units -> raw v_exp_f32.
#define QSCALE (0.125f * 1.44269504088896340736f)

static __device__ __forceinline__ bf16_t f2bf(float f) { return (bf16_t)f; }
static __device__ __forceinline__ bf16_t us2bf(unsigned short h) {
  return __builtin_bit_cast(bf16_t, h);
}
static __device__ __forceinline__ unsigned short bf2us(bf16_t b) {
  return __builtin_bit_cast(unsigned short, b);
}
static __device__ __forceinline__ float fexp2(float x) {
  return __builtin_amdgcn_exp2f(x);
}

// 16 consecutive, 16B-aligned floats -> v16bf (K-ascending fragment order)
static __device__ __forceinline__ v16bf cvt16(const float* p) {
  const float4* p4 = (const float4*)p;
  v16bf o;
#pragma unroll
  for (int i = 0; i < 4; ++i) {
    float4 f = p4[i];
    o[4*i+0] = f2bf(f.x); o[4*i+1] = f2bf(f.y);
    o[4*i+2] = f2bf(f.z); o[4*i+3] = f2bf(f.w);
  }
  return o;
}

__global__ __launch_bounds__(BLOCK)
void sdpa_fwd_kernel(const float* __restrict__ Q, const float* __restrict__ K,
                     const float* __restrict__ V, float* __restrict__ O,
                     float* __restrict__ A)
{
  // f32 V tile, row-major [kv][d] with padded stride (async-LDS destination)
  __shared__ float VTF[32 * VTS];
  // Per-wave P tile: 16 rows x 32 cols (padded rows, bf16)
  __shared__ unsigned short PL[WAVES][16 * PSTR];

  const int tid  = threadIdx.x;
  const int wid  = tid >> 5;
  const int lane = tid & 31;
  const int ln16 = lane & 15;
  const int half = lane >> 4;               // 0: lanes 0-15, 1: lanes 16-31

  const int bh    = blockIdx.x >> 5;        // head index (B*H = 64)
  const int qt    = blockIdx.x & 31;        // q tile (32 per head)
  const int qbase = qt * QTILE + wid * 16;  // first q row owned by this wave

  const size_t headoff = (size_t)bh * S_LEN * D_DIM;
  const float* Qh = Q + headoff;
  const float* Kh = K + headoff;
  const float* Vh = V + headoff;
  float*       Ah = A + (size_t)bh * S_LEN * S_LEN;

  // ---- Q fragments (A operand, 16x32 each), pre-scaled by log2e/temp ----
  // A layout (16-bit, 16x32): row M = lane%16; K: e -> kb+e (contiguous)
  const int ka   = half ? 8 : 0;
  const int kb   = half ? 16 : 0;
  const int qrow = qbase + ln16;
  v16bf qa0, qa1;
  {
    const float* qp = Qh + (size_t)qrow * D_DIM;
#pragma unroll
    for (int e = 0; e < 8; ++e) {
      qa0[e]     = f2bf(qp[      kb + e] * QSCALE);
      qa0[8 + e] = f2bf(qp[8 +   kb + e] * QSCALE);
      qa1[e]     = f2bf(qp[32 +  kb + e] * QSCALE);
      qa1[8 + e] = f2bf(qp[40 +  kb + e] * QSCALE);
    }
  }

  // Per-thread async-staging addresses for the V tile:
  // thread t copies 16 consecutive floats of row kv = t>>2, at d = (t&3)*16.
  const int sv_kv = tid >> 2;
  const int sv_d  = (tid & 3) * 16;
  const unsigned sv_lds = (unsigned)(size_t)&VTF[sv_kv * VTS + sv_d];

  // =======================  pass 1: softmax stats  ========================
  // C layout: VGPR r -> row r (lanes 0-15) / row r+8 (lanes 16-31), col = lane%16
  float m[8], l[8];
#pragma unroll
  for (int r = 0; r < 8; ++r) { m[r] = -3.0e38f; l[r] = 0.0f; }

  // two 16-wide tiles per iteration: one rescale-exp per row per 32 columns
  for (int kp = 0; kp < S_LEN / 32; ++kp) {
    const float* kr0 = Kh + (size_t)(kp * 32 + ln16) * D_DIM;
    const float* kr1 = kr0 + 16 * D_DIM;
    if (kp + 1 < S_LEN / 32)
      __builtin_prefetch(kr0 + 32 * D_DIM, 0, 3);
    v8f c0 = {}, c1 = {};
    {
      v16bf b0 = cvt16(kr0 + kb);
      v16bf b1 = cvt16(kr0 + 32 + kb);
      c0 = __builtin_amdgcn_wmma_f32_16x16x32_bf16(false, qa0, false, b0, (short)0, c0, false, false);
      c0 = __builtin_amdgcn_wmma_f32_16x16x32_bf16(false, qa1, false, b1, (short)0, c0, false, false);
    }
    {
      v16bf b0 = cvt16(kr1 + kb);
      v16bf b1 = cvt16(kr1 + 32 + kb);
      c1 = __builtin_amdgcn_wmma_f32_16x16x32_bf16(false, qa0, false, b0, (short)0, c1, false, false);
      c1 = __builtin_amdgcn_wmma_f32_16x16x32_bf16(false, qa1, false, b1, (short)0, c1, false, false);
    }
#pragma unroll
    for (int r = 0; r < 8; ++r) {
      float s0 = c0[r], s1 = c1[r];
      float mn = fmaxf(m[r], fmaxf(s0, s1));
      l[r] = l[r] * fexp2(m[r] - mn) + fexp2(s0 - mn) + fexp2(s1 - mn);
      m[r] = mn;
    }
  }
  // reduce (m,l) across the 16 lanes holding the same row (xor stays in half)
#pragma unroll
  for (int mk = 1; mk <= 8; mk <<= 1) {
#pragma unroll
    for (int r = 0; r < 8; ++r) {
      float mo = __shfl_xor(m[r], mk, 32);
      float lo = __shfl_xor(l[r], mk, 32);
      float mn = fmaxf(m[r], mo);
      l[r] = l[r] * fexp2(m[r] - mn) + lo * fexp2(mo - mn);
      m[r] = mn;
    }
  }
  float inv[8];
#pragma unroll
  for (int r = 0; r < 8; ++r) inv[r] = 1.0f / l[r];

  // =====================  pass 2: attn write + P@V  =======================
  v8f oacc[4] = {v8f{}, v8f{}, v8f{}, v8f{}};

  for (int kp = 0; kp < S_LEN / 32; ++kp) {
    const int kbase = kp * 32;

    __syncthreads();  // previous iteration's VTF readers are done
    {
      // Async-stage the V tile straight into LDS (no VGPR round-trip):
      // 4x GLOBAL_LOAD_ASYNC_TO_LDS_B128; the shared `offset:` immediate
      // advances both the global and the LDS address (ISA 08 §4.4).
      const float* gsrc = Vh + (size_t)(kbase + sv_kv) * D_DIM + sv_d;
      asm volatile("global_load_async_to_lds_b128 %0, %1, off offset:0"
                   :: "v"(sv_lds), "v"(gsrc) : "memory");
      asm volatile("global_load_async_to_lds_b128 %0, %1, off offset:16"
                   :: "v"(sv_lds), "v"(gsrc) : "memory");
      asm volatile("global_load_async_to_lds_b128 %0, %1, off offset:32"
                   :: "v"(sv_lds), "v"(gsrc) : "memory");
      asm volatile("global_load_async_to_lds_b128 %0, %1, off offset:48"
                   :: "v"(sv_lds), "v"(gsrc) : "memory");
      if (kp + 1 < S_LEN / 32)
        __builtin_prefetch(gsrc + 32 * D_DIM, 0, 3);
      asm volatile("s_wait_asynccnt 0" ::: "memory");
    }
    __syncthreads();  // publish the staged tile to all waves

    // two 16-wide score tiles -> p, write attn, stage P into LDS
#pragma unroll
    for (int t01 = 0; t01 < 2; ++t01) {
      const float* kr = Kh + (size_t)(kbase + t01 * 16 + ln16) * D_DIM;
      v16bf b0 = cvt16(kr + kb);
      v16bf b1 = cvt16(kr + 32 + kb);
      v8f c = {};
      c = __builtin_amdgcn_wmma_f32_16x16x32_bf16(false, qa0, false, b0, (short)0, c, false, false);
      c = __builtin_amdgcn_wmma_f32_16x16x32_bf16(false, qa1, false, b1, (short)0, c, false, false);
#pragma unroll
      for (int r = 0; r < 8; ++r) {
        float p  = fexp2(c[r] - m[r]) * inv[r];
        const int rl = r + half * 8;                 // local q row 0..15
        Ah[(size_t)(qbase + rl) * S_LEN + kbase + t01 * 16 + ln16] = p;
        PL[wid][rl * PSTR + t01 * 16 + ln16] = bf2us(f2bf(p));
      }
    }
    __syncthreads();  // PL (and uniformity) before re-reading as A operand

    // A operand: P (16x32) from padded LDS rows
    v16bf pa;
#pragma unroll
    for (int e = 0; e < 8; ++e) {
      pa[e]     = us2bf(PL[wid][ln16 * PSTR +      kb + e]);
      pa[8 + e] = us2bf(PL[wid][ln16 * PSTR + 8 +  kb + e]);
    }
    // B operands: V tile rows kb..kb+15, 4 chunks of 16 output columns
#pragma unroll
    for (int nch = 0; nch < 4; ++nch) {
      const int d = nch * 16 + ln16;
      v16bf vb;
#pragma unroll
      for (int e = 0; e < 16; ++e)
        vb[e] = f2bf(VTF[(kb + e) * VTS + d]);
      oacc[nch] = __builtin_amdgcn_wmma_f32_16x16x32_bf16(false, pa, false, vb, (short)0,
                                                          oacc[nch], false, false);
    }
  }

  // ---- write O [B,H,S,D] ----
#pragma unroll
  for (int nch = 0; nch < 4; ++nch) {
#pragma unroll
    for (int r = 0; r < 8; ++r) {
      const int rl = r + half * 8;
      O[headoff + (size_t)(qbase + rl) * D_DIM + nch * 16 + ln16] = oacc[nch][r];
    }
  }
}

extern "C" void kernel_launch(void* const* d_in, const int* in_sizes, int n_in,
                              void* d_out, int out_size, void* d_ws, size_t ws_size,
                              hipStream_t stream) {
  (void)in_sizes; (void)n_in; (void)out_size; (void)d_ws; (void)ws_size;
  const float* q = (const float*)d_in[0];
  const float* k = (const float*)d_in[1];
  const float* v = (const float*)d_in[2];
  float* out  = (float*)d_out;                                   // [B,H,S,D]
  float* attn = out + (size_t)NBH * S_LEN * D_DIM;               // [B,H,S,S]
  dim3 grid(NBH * (S_LEN / QTILE));                              // 64*32 = 2048
  sdpa_fwd_kernel<<<grid, BLOCK, 0, stream>>>(q, k, v, out, attn);
}
